// GINVNNoEdge_55886114456251
// MI455X (gfx1250) — matrix operand order repaired and checked by reading, not compiled, not run on hardware
//
#include <hip/hip_runtime.h>
#include <hip/hip_bf16.h>

#define HID    128
#define NFEAT  9
#define NVOC   128
#define NLAYER 5
#define BN_EPS 1e-5f

typedef __attribute__((ext_vector_type(16))) __bf16          v16bf;
typedef __attribute__((ext_vector_type(8)))  float           v8f;
typedef __attribute__((ext_vector_type(8)))  unsigned short  ush8;

union Frag { ush8 u[2]; v16bf v; };

__device__ __forceinline__ unsigned short f2bf(float f) {
  unsigned int u = __builtin_bit_cast(unsigned int, f);
  u += 0x7FFFu + ((u >> 16) & 1u);
  return (unsigned short)(u >> 16);
}

// ---------------- AtomEncoder: h[n,c] = sum_f emb[f, x[n,f], c] ----------------
__global__ void k_atom_encode(const int* __restrict__ x, const float* __restrict__ emb,
                              float* __restrict__ h) {
  int n = blockIdx.x, c = threadIdx.x;
  float s = 0.f;
#pragma unroll
  for (int f = 0; f < NFEAT; ++f) {
    int idx = x[n * NFEAT + f];
    s += emb[(f * NVOC + idx) * HID + c];
  }
  h[n * HID + c] = s;
}

// ---------------- z = (1+eps)*h ----------------
__global__ void k_scale(const float* __restrict__ h, const float* __restrict__ eps,
                        float* __restrict__ z, int total) {
  int i = blockIdx.x * blockDim.x + threadIdx.x;
  if (i < total) z[i] = (1.0f + eps[0]) * h[i];
}

// ---------------- edge scatter-add: z[dst] += h[src] (32 lanes/edge, float4) ----------------
__global__ void k_scatter(const int* __restrict__ src, const int* __restrict__ dst,
                          const float* __restrict__ h, float* __restrict__ z, int E) {
  int gid = blockIdx.x * blockDim.x + threadIdx.x;
  int e = gid >> 5;
  if (e >= E) return;
  int q = (gid & 31) * 4;
  int s = src[e], d = dst[e];
  const float4 v = *reinterpret_cast<const float4*>(h + (size_t)s * HID + q);
  float* p = z + (size_t)d * HID + q;
  atomicAdd(p + 0, v.x); atomicAdd(p + 1, v.y);
  atomicAdd(p + 2, v.z); atomicAdd(p + 3, v.w);
}

// ---------------- segment sum nodes -> graphs ----------------
__global__ void k_segnodes(const int* __restrict__ batch, const float* __restrict__ h,
                           float* __restrict__ g, int N) {
  int gid = blockIdx.x * blockDim.x + threadIdx.x;
  int n = gid >> 5;
  if (n >= N) return;
  int q = (gid & 31) * 4;
  int b = batch[n];
  const float4 v = *reinterpret_cast<const float4*>(h + (size_t)n * HID + q);
  float* p = g + (size_t)b * HID + q;
  atomicAdd(p + 0, v.x); atomicAdd(p + 1, v.y);
  atomicAdd(p + 2, v.z); atomicAdd(p + 3, v.w);
}

// ---------------- f32 -> bf16 convert ----------------
__global__ void k_cvt_bf16(const float* __restrict__ x, unsigned short* __restrict__ y, int total) {
  int i = blockIdx.x * blockDim.x + threadIdx.x;
  if (i < total) y[i] = f2bf(x[i]);
}

// ---------------- pack W[k,j] -> Wt[j,k] as bf16 ----------------
__global__ void k_pack_w(const float* __restrict__ W, unsigned short* __restrict__ Wt) {
  int i = blockIdx.x * blockDim.x + threadIdx.x; // 0..HID*HID-1
  int k = i & (HID - 1), j = i >> 7;
  Wt[j * HID + k] = f2bf(W[k * HID + j]);
}

// ---------------- WMMA GEMM: out[M,128] = A_bf16[M,128] @ W + bias ----------------
// A row-major bf16, Bt = W^T row-major bf16 (Bt[j,k]). M multiple of 16.
// Block = 256 threads = 8 waves; wave w computes 16-row x 16-col tile (cols 16w..16w+15).
__global__ void k_gemm_wmma(const unsigned short* __restrict__ A,
                            const unsigned short* __restrict__ Bt,
                            const float* __restrict__ bias,
                            float* __restrict__ out, int M) {
  const int wave = threadIdx.x >> 5;
  const int lane = threadIdx.x & 31;
  const int r    = lane & 15;
  const int hi   = lane >> 4;
  const int rowBase = blockIdx.x * 16;
  const int colBase = wave * 16;

  v8f acc;
  const float bc = bias[colBase + r];
#pragma unroll
  for (int i = 0; i < 8; ++i) acc[i] = bc;

  const unsigned short* arow = A  + (size_t)(rowBase + r) * HID;
  const unsigned short* brow = Bt + (size_t)(colBase + r) * HID;

#pragma unroll
  for (int ks = 0; ks < 4; ++ks) {
    Frag fa, fb;
    // A 16x32 bf16 layout: lanes 0-15 hold K {0..7,16..23}, lanes 16-31 K {8..15,24..31}
    int ka = ks * 32 + hi * 8;
    fa.u[0] = *reinterpret_cast<const ush8*>(arow + ka);
    fa.u[1] = *reinterpret_cast<const ush8*>(arow + ka + 16);
    // B 32x16 bf16 layout: lanes 0-15 hold K {0..7,16..23}, lanes 16-31 K {8..15,24..31}
    int kb = ks * 32 + hi * 8;
    fb.u[0] = *reinterpret_cast<const ush8*>(brow + kb);
    fb.u[1] = *reinterpret_cast<const ush8*>(brow + kb + 16);
    acc = __builtin_amdgcn_wmma_f32_16x16x32_bf16(false, fa.v, false, fb.v,
                                                  (short)0, acc, false, false);
  }

  // D layout: VGPR i -> row (i + 8*hi), col = colBase + (lane&15)
#pragma unroll
  for (int i = 0; i < 8; ++i) {
    int row = rowBase + hi * 8 + i;
    out[(size_t)row * HID + colBase + r] = acc[i];
  }
}

// ---------------- column sum / sumsq (training-mode BN stats) ----------------
__global__ void k_colstats(const float* __restrict__ x, float* __restrict__ stats, int M) {
  int c  = threadIdx.x;            // 128 threads
  int r0 = blockIdx.x * 256;
  int r1 = r0 + 256; if (r1 > M) r1 = M;
  float s = 0.f, s2 = 0.f;
  for (int r = r0; r < r1; ++r) {
    float v = x[(size_t)r * HID + c];
    s += v; s2 += v * v;
  }
  atomicAdd(&stats[c], s);
  atomicAdd(&stats[HID + c], s2);
}

__global__ void k_stats_fin(float* __restrict__ stats, int M) {
  int c = threadIdx.x;
  float inv = 1.0f / (float)M;
  float mean = stats[c] * inv;
  float var  = stats[HID + c] * inv - mean * mean;
  stats[c]       = mean;
  stats[HID + c] = rsqrtf(var + BN_EPS);
}

// ---------------- BN apply (+optional relu), write f32 and/or bf16 ----------------
__global__ void k_bn(const float* __restrict__ x, const float* __restrict__ stats,
                     const float* __restrict__ g, const float* __restrict__ b,
                     float* __restrict__ outf, unsigned short* __restrict__ outb,
                     int total, int relu) {
  int i = blockIdx.x * blockDim.x + threadIdx.x;
  if (i >= total) return;
  int c = i & (HID - 1);
  float v = g[c] * (x[i] - stats[c]) * stats[HID + c] + b[c];
  if (relu) v = fmaxf(v, 0.f);
  if (outf) outf[i] = v;
  if (outb) outb[i] = f2bf(v);
}

// ---------------- y += x ----------------
__global__ void k_add(float* __restrict__ y, const float* __restrict__ x, int total) {
  int i = blockIdx.x * blockDim.x + threadIdx.x;
  if (i < total) y[i] += x[i];
}

// ---------------- h[n,c] += vn[batch[n],c] ----------------
__global__ void k_add_bcast(float* __restrict__ h, const float* __restrict__ vn,
                            const int* __restrict__ batch, int total) {
  int i = blockIdx.x * blockDim.x + threadIdx.x;
  if (i >= total) return;
  int n = i >> 7, c = i & (HID - 1);
  h[i] += vn[(size_t)batch[n] * HID + c];
}

// ---------------- broadcast single row to G rows ----------------
__global__ void k_bcast_row(float* __restrict__ y, const float* __restrict__ row, int total) {
  int i = blockIdx.x * blockDim.x + threadIdx.x;
  if (i < total) y[i] = row[i & (HID - 1)];
}

// ---------------- classifier tail: out[g] = sum_c relu(t[g,c])*w2[c] + b2 ----------------
__global__ void k_clf2(const float* __restrict__ t, const float* __restrict__ w2,
                       const float* __restrict__ b2, float* __restrict__ out) {
  __shared__ float red[HID];
  int g = blockIdx.x, c = threadIdx.x;
  red[c] = fmaxf(t[(size_t)g * HID + c], 0.f) * w2[c];
  __syncthreads();
  for (int s = 64; s > 0; s >>= 1) {
    if (c < s) red[c] += red[c + s];
    __syncthreads();
  }
  if (c == 0) out[g] = red[0] + b2[0];
}

extern "C" void kernel_launch(void* const* d_in, const int* in_sizes, int n_in,
                              void* d_out, int out_size, void* d_ws, size_t ws_size,
                              hipStream_t stream) {
  const int N = in_sizes[0] / NFEAT;      // 50000
  const int E = in_sizes[1] / 2;          // 800000
  const int G = out_size;                 // 2048

  const int*   x       = (const int*)d_in[0];
  const int*   src     = (const int*)d_in[1];
  const int*   dst     = (const int*)d_in[1] + E;
  const int*   batch   = (const int*)d_in[3];
  const float* atomemb = (const float*)d_in[4];
  const float* eps     = (const float*)d_in[5];
  const float* convW1  = (const float*)d_in[6];
  const float* convb1  = (const float*)d_in[7];
  const float* convg1  = (const float*)d_in[8];
  const float* convbt1 = (const float*)d_in[9];
  const float* convW2  = (const float*)d_in[10];
  const float* convb2  = (const float*)d_in[11];
  const float* bng     = (const float*)d_in[12];
  const float* bnb     = (const float*)d_in[13];
  const float* vn_emb  = (const float*)d_in[14];
  const float* vnW1    = (const float*)d_in[15];
  const float* vnb1    = (const float*)d_in[16];
  const float* vng1    = (const float*)d_in[17];
  const float* vnbt1   = (const float*)d_in[18];
  const float* vnW2    = (const float*)d_in[19];
  const float* vnb2    = (const float*)d_in[20];
  const float* clfW1   = (const float*)d_in[21];
  const float* clfb1   = (const float*)d_in[22];
  const float* clfW2   = (const float*)d_in[23];
  const float* clfb2   = (const float*)d_in[24];
  float*       out     = (float*)d_out;

  // ---- workspace bump allocator ----
  char*  ws  = (char*)d_ws;
  size_t off = 0;
  auto alloc = [&](size_t bytes) -> void* {
    void* p = ws + off;
    off = (off + bytes + 255) & ~(size_t)255;
    return p;
  };
  const size_t NH = (size_t)N * HID, GH = (size_t)G * HID;
  float*          zf    = (float*)alloc(NH * 4);
  float*          z2f   = (float*)alloc(NH * 4);
  float*          hf    = (float*)alloc(NH * 4);
  unsigned short* actb  = (unsigned short*)alloc(NH * 2);
  unsigned short* wtb   = (unsigned short*)alloc((size_t)HID * HID * 2);
  float*          stats = (float*)alloc(2 * HID * 4);
  float*          vtf   = (float*)alloc(GH * 4);
  float*          vzf   = (float*)alloc(GH * 4);
  float*          vnh   = (float*)alloc(GH * 4);
  unsigned short* vactb = (unsigned short*)alloc(GH * 2);
  (void)ws_size;

  const int TB = 256;
  auto blocks = [](size_t total, int tb) { return (unsigned)((total + tb - 1) / tb); };

  auto bn_stats = [&](const float* xbuf, int M) {
    hipMemsetAsync(stats, 0, 2 * HID * sizeof(float), stream);
    k_colstats<<<blocks(M, 256), HID, 0, stream>>>(xbuf, stats, M);
    k_stats_fin<<<1, HID, 0, stream>>>(stats, M);
  };

  // 1) AtomEncoder
  k_atom_encode<<<N, HID, 0, stream>>>(x, atomemb, hf);
  // 2) virtual-node state init: broadcast vn_emb row to G rows
  k_bcast_row<<<blocks(GH, TB), TB, 0, stream>>>(vnh, vn_emb, (int)GH);

  for (int l = 0; l < NLAYER; ++l) {
    // GIN aggregate: z = (1+eps_l)*h + scatter_add(h[src] -> dst)
    k_scale<<<blocks(NH, TB), TB, 0, stream>>>(hf, eps + l, zf, (int)NH);
    k_scatter<<<blocks((size_t)E * 32, TB), TB, 0, stream>>>(src, dst, hf, zf, E);

    // MLP1: bn(relu later) — z2 = z @ W1 + b1
    k_cvt_bf16<<<blocks(NH, TB), TB, 0, stream>>>(zf, actb, (int)NH);
    k_pack_w<<<(HID * HID) / TB, TB, 0, stream>>>(convW1 + (size_t)l * HID * HID, wtb);
    k_gemm_wmma<<<N / 16, 256, 0, stream>>>(actb, wtb, convb1 + l * HID, z2f, N);
    bn_stats(z2f, N);
    k_bn<<<blocks(NH, TB), TB, 0, stream>>>(z2f, stats, convg1 + l * HID, convbt1 + l * HID,
                                            nullptr, actb, (int)NH, 1);

    // MLP2: z3 = relu(bn(z2)) @ W2 + b2 ; h = bn(z3) [+relu if l<4]
    k_pack_w<<<(HID * HID) / TB, TB, 0, stream>>>(convW2 + (size_t)l * HID * HID, wtb);
    k_gemm_wmma<<<N / 16, 256, 0, stream>>>(actb, wtb, convb2 + l * HID, zf, N);
    bn_stats(zf, N);
    k_bn<<<blocks(NH, TB), TB, 0, stream>>>(zf, stats, bng + l * HID, bnb + l * HID,
                                            hf, nullptr, (int)NH, (l < NLAYER - 1) ? 1 : 0);

    if (l < NLAYER - 1) {
      // vt = segment_sum(h, batch) + vn_h
      hipMemsetAsync(vtf, 0, GH * sizeof(float), stream);
      k_segnodes<<<blocks((size_t)N * 32, TB), TB, 0, stream>>>(batch, hf, vtf, N);
      k_add<<<blocks(GH, TB), TB, 0, stream>>>(vtf, vnh, (int)GH);
      // vn MLP: vz = relu(bn(vt@vnW1+vnb1)); vn_h += vz@vnW2 + vnb2
      k_cvt_bf16<<<blocks(GH, TB), TB, 0, stream>>>(vtf, vactb, (int)GH);
      k_pack_w<<<(HID * HID) / TB, TB, 0, stream>>>(vnW1 + (size_t)l * HID * HID, wtb);
      k_gemm_wmma<<<G / 16, 256, 0, stream>>>(vactb, wtb, vnb1 + l * HID, vzf, G);
      bn_stats(vzf, G);
      k_bn<<<blocks(GH, TB), TB, 0, stream>>>(vzf, stats, vng1 + l * HID, vnbt1 + l * HID,
                                              nullptr, vactb, (int)GH, 1);
      k_pack_w<<<(HID * HID) / TB, TB, 0, stream>>>(vnW2 + (size_t)l * HID * HID, wtb);
      k_gemm_wmma<<<G / 16, 256, 0, stream>>>(vactb, wtb, vnb2 + l * HID, vzf, G);
      k_add<<<blocks(GH, TB), TB, 0, stream>>>(vnh, vzf, (int)GH);
      // h += vn_h[batch]
      k_add_bcast<<<blocks(NH, TB), TB, 0, stream>>>(hf, vnh, batch, (int)NH);
    }
  }

  // readout + classifier
  hipMemsetAsync(vtf, 0, GH * sizeof(float), stream);
  k_segnodes<<<blocks((size_t)N * 32, TB), TB, 0, stream>>>(batch, hf, vtf, N);
  k_cvt_bf16<<<blocks(GH, TB), TB, 0, stream>>>(vtf, vactb, (int)GH);
  k_pack_w<<<(HID * HID) / TB, TB, 0, stream>>>(clfW1, wtb);
  k_gemm_wmma<<<G / 16, 256, 0, stream>>>(vactb, wtb, clfb1, vzf, G);
  k_clf2<<<G, HID, 0, stream>>>(vzf, clfW2, clfb2, out);
}
